// Llama4TextBlock_9483287789693
// MI455X (gfx1250) — matrix operand order repaired and checked by reading, not compiled, not run on hardware
//
#include <hip/hip_runtime.h>

#define H_DIM 1024
#define I_DIM 2048
#define E_NUM 8
#define T_TOK 8192
#define MTILE 32                       // rows per GEMM block tile
#define PT_TOK (T_TOK + MTILE * E_NUM) // permuted rows incl. per-expert padding
#define MAXTILES (PT_TOK / MTILE)      // 264
#define LN_EPS 1e-5f

typedef __bf16 bf16;
typedef __bf16 v16bf __attribute__((ext_vector_type(16)));
typedef __bf16 v8bf  __attribute__((ext_vector_type(8)));
typedef __bf16 v4bf  __attribute__((ext_vector_type(4)));
typedef float  v8f   __attribute__((ext_vector_type(8)));

// ---------------- WMMA fragment helpers (CDNA5 16x16x32 bf16 layouts) ------
// A (16x32, 16-bit): lane L holds row M=L%16; K = c0..c0+7 and c0+16..c0+23,
// c0 = (L<16 ? 0 : 8).  Two contiguous 16B loads -> adjacent VGPRs.
__device__ __forceinline__ v16bf load_a_frag(const bf16* __restrict__ row,
                                             int k0, int lane) {
  const int c0 = k0 + ((lane & 16) ? 8 : 0);
  v8bf lo = *(const v8bf*)(row + c0);
  v8bf hi = *(const v8bf*)(row + c0 + 16);
  v16bf f;
#pragma unroll
  for (int i = 0; i < 8; ++i) { f[i] = lo[i]; f[i + 8] = hi[i]; }
  return f;
}

// B (32x16, 16-bit): lane L holds column N=L%16; K = c0..c0+7 and c0+16..c0+23,
// c0 = (L<16 ? 0 : 8).  Two contiguous 16B loads (weights stored transposed).
__device__ __forceinline__ v16bf load_b_frag(const bf16* __restrict__ col,
                                             int k0, int lane) {
  const int c0 = k0 + ((lane & 16) ? 8 : 0);
  v8bf lo = *(const v8bf*)(col + c0);
  v8bf hi = *(const v8bf*)(col + c0 + 16);
  v16bf f;
#pragma unroll
  for (int i = 0; i < 8; ++i) { f[i] = lo[i]; f[i + 8] = hi[i]; }
  return f;
}

__device__ __forceinline__ v8f zero_acc() {
  v8f c;
#pragma unroll
  for (int i = 0; i < 8; ++i) c[i] = 0.0f;
  return c;
}

__device__ __forceinline__ v8f wmma_bf16(v16bf a, v16bf b, v8f c) {
  return __builtin_amdgcn_wmma_f32_16x16x32_bf16(false, a, false, b,
                                                 (short)0, c, false, false);
}

// ---------------- K1: fp32 -> bf16 transpose-convert (LDS tiled) -----------
__global__ __launch_bounds__(256) void transpose_cvt_kernel(
    const float* __restrict__ in, bf16* __restrict__ out, int R, int C) {
  __shared__ float tile[32][33];
  const size_t mat = (size_t)blockIdx.z * (size_t)R * (size_t)C;
  const float* src = in + mat;
  bf16* dst = out + mat;
  const int c0 = blockIdx.x * 32, r0 = blockIdx.y * 32;
  const int tx = threadIdx.x, ty = threadIdx.y;   // 32 x 8
#pragma unroll
  for (int i = 0; i < 32; i += 8)
    tile[ty + i][tx] = src[(size_t)(r0 + ty + i) * C + (c0 + tx)];
  __syncthreads();
#pragma unroll
  for (int i = 0; i < 32; i += 8)
    dst[(size_t)(c0 + ty + i) * R + (r0 + tx)] = (bf16)tile[tx][ty + i];
}

// ---------------- K2: LayerNorm + router (one block per token) -------------
__global__ __launch_bounds__(256) void ln_router_kernel(
    const float* __restrict__ X, const float* __restrict__ lnw,
    const float* __restrict__ lnb, const float* __restrict__ rw,
    bf16* __restrict__ xb, bf16* __restrict__ xs,
    int* __restrict__ top_idx, int* __restrict__ counts) {
  const int t = blockIdx.x, tid = threadIdx.x;
  const float* xr = X + (size_t)t * H_DIM;
  float4 v = *(const float4*)(xr + tid * 4);

  __shared__ float ssum[256], ssq[256];
  ssum[tid] = v.x + v.y + v.z + v.w;
  ssq[tid]  = v.x * v.x + v.y * v.y + v.z * v.z + v.w * v.w;
  __syncthreads();
  for (int s = 128; s > 0; s >>= 1) {
    if (tid < s) { ssum[tid] += ssum[tid + s]; ssq[tid] += ssq[tid + s]; }
    __syncthreads();
  }
  __shared__ float stat[2];
  if (tid == 0) {
    float mu = ssum[0] * (1.0f / H_DIM);
    stat[0] = mu;
    stat[1] = rsqrtf(ssq[0] * (1.0f / H_DIM) - mu * mu + LN_EPS);
  }
  __syncthreads();
  const float mu = stat[0], rs = stat[1];

  const int h0 = tid * 4;
  float vv[4] = {v.x, v.y, v.z, v.w};
  float nv[4];
  v4bf ob;
#pragma unroll
  for (int j = 0; j < 4; ++j) {
    nv[j] = (vv[j] - mu) * rs * lnw[h0 + j] + lnb[h0 + j];
    ob[j] = (bf16)nv[j];
  }
  *(v4bf*)(xb + (size_t)t * H_DIM + h0) = ob;

  // router logits: 8 dot-products of length H
  float acc[E_NUM];
#pragma unroll
  for (int e = 0; e < E_NUM; ++e) acc[e] = 0.0f;
#pragma unroll
  for (int j = 0; j < 4; ++j) {
    const float* r = rw + (size_t)(h0 + j) * E_NUM;
#pragma unroll
    for (int e = 0; e < E_NUM; ++e) acc[e] += nv[j] * r[e];
  }
  __shared__ float sl[256][E_NUM + 1];
#pragma unroll
  for (int e = 0; e < E_NUM; ++e) sl[tid][e] = acc[e];
  __syncthreads();
  for (int s = 128; s > 0; s >>= 1) {
    if (tid < s) {
#pragma unroll
      for (int e = 0; e < E_NUM; ++e) sl[tid][e] += sl[tid + s][e];
    }
    __syncthreads();
  }
  __shared__ float sscore;
  if (tid == 0) {
    int be = 0;
    float bl = sl[0][0];
#pragma unroll
    for (int e = 1; e < E_NUM; ++e)
      if (sl[0][e] > bl) { bl = sl[0][e]; be = e; }
    sscore = 1.0f / (1.0f + __expf(-bl));
    top_idx[t] = be;
    atomicAdd(&counts[be], 1);
  }
  __syncthreads();
  const float sc = sscore;
  v4bf os;
#pragma unroll
  for (int j = 0; j < 4; ++j) os[j] = (bf16)(nv[j] * sc);
  *(v4bf*)(xs + (size_t)t * H_DIM + h0) = os;
}

// ---------------- K3: expert offsets (padded to MTILE) + perm init ---------
__global__ void setup_meta_kernel(const int* __restrict__ counts,
                                  int* __restrict__ off,
                                  int* __restrict__ tile_expert,
                                  int* __restrict__ ntiles,
                                  int* __restrict__ perm) {
  const int tid = threadIdx.x;
  for (int i = tid; i < PT_TOK; i += blockDim.x) perm[i] = -1;
  if (tid == 0) {
    int o = 0;
    for (int e = 0; e < E_NUM; ++e) {
      off[e] = o;
      int te = (counts[e] + MTILE - 1) / MTILE;
      for (int k = 0; k < te; ++k) tile_expert[(o / MTILE) + k] = e;
      o += te * MTILE;
    }
    off[E_NUM] = o;
    *ntiles = o / MTILE;
  }
}

// ---------------- K4: gather tokens by expert ------------------------------
__global__ void gather_kernel(const int* __restrict__ top_idx,
                              const int* __restrict__ off,
                              int* __restrict__ cursor,
                              int* __restrict__ perm) {
  const int t = blockIdx.x * blockDim.x + threadIdx.x;
  if (t >= T_TOK) return;
  const int e = top_idx[t];
  const int p = atomicAdd(&cursor[e], 1);
  perm[off[e] + p] = t;
}

// ---------------- K5/K7: fused gate+up GEMM + SiLU*mul ---------------------
// Block: 8 waves, tile MTILE(32) rows x 128 cols of I.  Each wave owns two
// 16x16 C subtiles per matrix, reusing B fragments: 4 WMMAs per 128B loaded.
// Software-pipelined: next K-chunk's 8 fragment loads are issued before the
// current chunk's WMMAs, hiding load latency behind matrix issue.  The wrap
// (k+32)&(K-1) makes the last iteration re-load chunk 0 (discarded) so the
// loop stays guard-free and unrolls into pure register renaming.
__global__ __launch_bounds__(256) void moe_gate_up_kernel(
    const bf16* __restrict__ X, const bf16* __restrict__ Gt,
    const bf16* __restrict__ Ut, bf16* __restrict__ Hm,
    const bf16* __restrict__ zrow, const int* __restrict__ perm,
    const int* __restrict__ tile_expert, const int* __restrict__ ntiles) {
  const int tile = blockIdx.x;
  size_t wstride = 0;
  if (tile_expert) {
    if (tile >= *ntiles) return;
    wstride = (size_t)tile_expert[tile] * (size_t)H_DIM * (size_t)I_DIM;
  }
  const int tid = threadIdx.x, lane = tid & 31, w = tid >> 5;
  const int mr = lane & 15;
  const int r0 = tile * MTILE + mr;
  const int r1 = r0 + 16;
  const int tok0 = perm ? perm[r0] : r0;
  const int tok1 = perm ? perm[r1] : r1;
  const bf16* A0 = (tok0 >= 0) ? (X + (size_t)tok0 * H_DIM) : zrow;
  const bf16* A1 = (tok1 >= 0) ? (X + (size_t)tok1 * H_DIM) : zrow;
  const int n = blockIdx.y * 128 + w * 16 + mr;
  const bf16* Bg = Gt + wstride + (size_t)n * H_DIM;
  const bf16* Bu = Ut + wstride + (size_t)n * H_DIM;

  v8f cg0 = zero_acc(), cg1 = zero_acc();
  v8f cu0 = zero_acc(), cu1 = zero_acc();

  v16bf a0 = load_a_frag(A0, 0, lane);
  v16bf a1 = load_a_frag(A1, 0, lane);
  v16bf bg = load_b_frag(Bg, 0, lane);
  v16bf bu = load_b_frag(Bu, 0, lane);
#pragma unroll 4
  for (int k0 = 0; k0 < H_DIM; k0 += 32) {
    const int kn = (k0 + 32) & (H_DIM - 1);
    v16bf na0 = load_a_frag(A0, kn, lane);
    v16bf na1 = load_a_frag(A1, kn, lane);
    v16bf nbg = load_b_frag(Bg, kn, lane);
    v16bf nbu = load_b_frag(Bu, kn, lane);
    cg0 = wmma_bf16(a0, bg, cg0);
    cg1 = wmma_bf16(a1, bg, cg1);
    cu0 = wmma_bf16(a0, bu, cu0);
    cu1 = wmma_bf16(a1, bu, cu1);
    a0 = na0; a1 = na1; bg = nbg; bu = nbu;
  }

  // C layout: lane holds col n=lane%16, rows (lane<16?0:8)+r
  __shared__ bf16 st[MTILE][128];
  const int cmb = (lane & 16) ? 8 : 0;
  const int cn = w * 16 + mr;
#pragma unroll
  for (int r = 0; r < 8; ++r) {
    float g0 = cg0[r], g1 = cg1[r];
    st[cmb + r][cn]      = (bf16)((g0 / (1.0f + __expf(-g0))) * cu0[r]);
    st[16 + cmb + r][cn] = (bf16)((g1 / (1.0f + __expf(-g1))) * cu1[r]);
  }
  __syncthreads();
  const int trow = tid >> 3, tc = (tid & 7) * 16;   // 32 rows x 128 cols
  const size_t dst = (size_t)(tile * MTILE + trow) * I_DIM + blockIdx.y * 128 + tc;
  *(v8bf*)(Hm + dst)     = *(const v8bf*)&st[trow][tc];
  *(v8bf*)(Hm + dst + 8) = *(const v8bf*)&st[trow][tc + 8];
}

// ---------------- K6/K8: down-proj GEMM + residual/accumulate --------------
// Dt is [H][I] bf16 (transposed).  mode 0: out = hidden + c (shared pass),
// mode 1: out += c (routed pass; token permutation is bijective => race-free).
__global__ __launch_bounds__(256) void moe_down_kernel(
    const bf16* __restrict__ Hm, const bf16* __restrict__ Dt,
    const float* __restrict__ hidden, float* __restrict__ Out,
    const int* __restrict__ perm, const int* __restrict__ tile_expert,
    const int* __restrict__ ntiles, int accumulate) {
  const int tile = blockIdx.x;
  size_t wstride = 0;
  if (tile_expert) {
    if (tile >= *ntiles) return;
    wstride = (size_t)tile_expert[tile] * (size_t)H_DIM * (size_t)I_DIM;
  }
  const int tid = threadIdx.x, lane = tid & 31, w = tid >> 5;
  const int mr = lane & 15;
  const bf16* A0 = Hm + (size_t)(tile * MTILE + mr) * I_DIM;
  const bf16* A1 = A0 + (size_t)16 * I_DIM;
  const int n = blockIdx.y * 128 + w * 16 + mr;
  const bf16* Bd = Dt + wstride + (size_t)n * I_DIM;

  v8f c0 = zero_acc(), c1 = zero_acc();

  v16bf a0 = load_a_frag(A0, 0, lane);
  v16bf a1 = load_a_frag(A1, 0, lane);
  v16bf b  = load_b_frag(Bd, 0, lane);
#pragma unroll 4
  for (int k0 = 0; k0 < I_DIM; k0 += 32) {
    const int kn = (k0 + 32) & (I_DIM - 1);
    v16bf na0 = load_a_frag(A0, kn, lane);
    v16bf na1 = load_a_frag(A1, kn, lane);
    v16bf nb  = load_b_frag(Bd, kn, lane);
    c0 = wmma_bf16(a0, b, c0);
    c1 = wmma_bf16(a1, b, c1);
    a0 = na0; a1 = na1; b = nb;
  }

  __shared__ float st[MTILE][128];
  const int cmb = (lane & 16) ? 8 : 0;
  const int cn = w * 16 + mr;
#pragma unroll
  for (int r = 0; r < 8; ++r) {
    st[cmb + r][cn]      = c0[r];
    st[16 + cmb + r][cn] = c1[r];
  }
  __syncthreads();

  const int trow = tid >> 3, tc = (tid & 7) * 16;   // 32 rows x 128 cols
  const int prow = tile * MTILE + trow;
  const int tok = perm ? perm[prow] : prow;
  if (tok >= 0) {
    const size_t base = (size_t)tok * H_DIM + blockIdx.y * 128 + tc;
#pragma unroll
    for (int q = 0; q < 4; ++q) {
      float4 cv = *(const float4*)&st[trow][tc + q * 4];
      float4* op = (float4*)(Out + base + q * 4);
      float4 rv;
      if (accumulate) rv = *op;
      else            rv = *(const float4*)(hidden + base + q * 4);
      rv.x += cv.x; rv.y += cv.y; rv.z += cv.z; rv.w += cv.w;
      *op = rv;
    }
  }
}

// ---------------------------- host launcher --------------------------------
extern "C" void kernel_launch(void* const* d_in, const int* in_sizes, int n_in,
                              void* d_out, int out_size, void* d_ws,
                              size_t ws_size, hipStream_t stream) {
  (void)in_sizes; (void)n_in; (void)out_size; (void)ws_size;
  const float* hidden = (const float*)d_in[0];
  const float* ln_w   = (const float*)d_in[1];
  const float* ln_b   = (const float*)d_in[2];
  const float* rw     = (const float*)d_in[3];
  const float* gate_w = (const float*)d_in[4];
  const float* up_w   = (const float*)d_in[5];
  const float* down_w = (const float*)d_in[6];
  const float* sh_g   = (const float*)d_in[7];
  const float* sh_u   = (const float*)d_in[8];
  const float* sh_d   = (const float*)d_in[9];
  float* out = (float*)d_out;

  // workspace carve-up
  char* p = (char*)d_ws;
  auto take = [&](size_t bytes) -> char* {
    char* r = p;
    p += (bytes + 255) & ~(size_t)255;
    return r;
  };
  const size_t EHI = (size_t)E_NUM * H_DIM * I_DIM;
  const size_t HI  = (size_t)H_DIM * I_DIM;
  bf16* Gb   = (bf16*)take(EHI * 2);                 // [E][I][H]
  bf16* Ub   = (bf16*)take(EHI * 2);                 // [E][I][H]
  bf16* Db   = (bf16*)take(EHI * 2);                 // [E][H][I]
  bf16* SGb  = (bf16*)take(HI * 2);                  // [I][H]
  bf16* SUb  = (bf16*)take(HI * 2);                  // [I][H]
  bf16* SDb  = (bf16*)take(HI * 2);                  // [H][I]
  bf16* xb   = (bf16*)take((size_t)T_TOK * H_DIM * 2);
  bf16* xs   = (bf16*)take((size_t)T_TOK * H_DIM * 2);
  bf16* hmid = (bf16*)take((size_t)PT_TOK * I_DIM * 2);
  bf16* zrow = (bf16*)take((size_t)H_DIM * 2);       // zeroed dummy A row
  int* top_idx = (int*)take((size_t)T_TOK * 4);
  int* perm    = (int*)take((size_t)PT_TOK * 4);
  int* meta    = (int*)take((size_t)(26 + MAXTILES) * 4);
  int* counts      = meta;            // 8
  int* cursor      = meta + 8;        // 8
  int* off         = meta + 16;       // 9
  int* ntiles      = meta + 25;       // 1
  int* tile_expert = meta + 26;       // MAXTILES

  hipMemsetAsync(meta, 0, (size_t)(26 + MAXTILES) * 4, stream);
  hipMemsetAsync(zrow, 0, (size_t)H_DIM * 2, stream);

  dim3 tb(32, 8, 1);
  transpose_cvt_kernel<<<dim3(I_DIM / 32, H_DIM / 32, E_NUM), tb, 0, stream>>>(
      gate_w, Gb, H_DIM, I_DIM);
  transpose_cvt_kernel<<<dim3(I_DIM / 32, H_DIM / 32, E_NUM), tb, 0, stream>>>(
      up_w, Ub, H_DIM, I_DIM);
  transpose_cvt_kernel<<<dim3(H_DIM / 32, I_DIM / 32, E_NUM), tb, 0, stream>>>(
      down_w, Db, I_DIM, H_DIM);
  transpose_cvt_kernel<<<dim3(I_DIM / 32, H_DIM / 32, 1), tb, 0, stream>>>(
      sh_g, SGb, H_DIM, I_DIM);
  transpose_cvt_kernel<<<dim3(I_DIM / 32, H_DIM / 32, 1), tb, 0, stream>>>(
      sh_u, SUb, H_DIM, I_DIM);
  transpose_cvt_kernel<<<dim3(H_DIM / 32, I_DIM / 32, 1), tb, 0, stream>>>(
      sh_d, SDb, I_DIM, H_DIM);

  ln_router_kernel<<<T_TOK, 256, 0, stream>>>(hidden, ln_w, ln_b, rw, xb, xs,
                                              top_idx, counts);
  setup_meta_kernel<<<1, 256, 0, stream>>>(counts, off, tile_expert, ntiles,
                                           perm);
  gather_kernel<<<T_TOK / 256, 256, 0, stream>>>(top_idx, off, cursor, perm);

  // shared expert: out = hidden + shared_swiglu(xb)
  moe_gate_up_kernel<<<dim3(T_TOK / MTILE, I_DIM / 128), 256, 0, stream>>>(
      xb, SGb, SUb, hmid, zrow, nullptr, nullptr, nullptr);
  moe_down_kernel<<<dim3(T_TOK / MTILE, H_DIM / 128), 256, 0, stream>>>(
      hmid, SDb, hidden, out, nullptr, nullptr, nullptr, 0);

  // routed experts: out += expert_swiglu(xs) gathered by expert
  moe_gate_up_kernel<<<dim3(MAXTILES, I_DIM / 128), 256, 0, stream>>>(
      xs, Gb, Ub, hmid, zrow, perm, tile_expert, ntiles);
  moe_down_kernel<<<dim3(MAXTILES, H_DIM / 128), 256, 0, stream>>>(
      hmid, Db, hidden, out, perm, tile_expert, ntiles, 1);
}
